// SiamTwoStage_76716705841929
// MI455X (gfx1250) — matrix operand order repaired and checked by reading, not compiled, not run on hardware
//
#include <hip/hip_runtime.h>
#include <math.h>

// ---------------------------------------------------------------------------
// CDNA5 (gfx1250) WMMA types
// ---------------------------------------------------------------------------
typedef __attribute__((ext_vector_type(16))) __bf16 v16bf;
typedef __attribute__((ext_vector_type(8)))  float  v8f;

__device__ __forceinline__ float sigmoidf_(float x) {
    return 1.0f / (1.0f + expf(-x));
}

// ---------------------------------------------------------------------------
// NCHW fp32 -> zero-padded NHWC bf16: xh[b][H+2][W+2][C], halo = 0.
// Makes the conv inner loop completely branch-free.
// ---------------------------------------------------------------------------
__global__ void k_to_nhwc_pad(const float* __restrict__ x, __bf16* __restrict__ xh,
                              int Bn, int C, int H, int W)
{
    int Hp = H + 2, Wp = W + 2;
    int idx = blockIdx.x * blockDim.x + threadIdx.x;
    int total = Bn * Hp * Wp * C;
    if (idx >= total) return;
    int c  = idx % C;
    int t  = idx / C;
    int xp = t % Wp; t /= Wp;
    int yp = t % Hp;
    int b  = t / Hp;
    int iy = yp - 1, ix = xp - 1;
    float v = 0.0f;
    if (iy >= 0 && iy < H && ix >= 0 && ix < W)
        v = x[((size_t)(b * C + c) * H + iy) * W + ix];
    xh[idx] = (__bf16)v;
}

// Fused concat([roi_s, tpl3g[ri/K]], channel) -> zero-padded NHWC bf16.
__global__ void k_cat_nhwc_pad(const float* __restrict__ roi_s, const float* __restrict__ tplg,
                               __bf16* __restrict__ xh, int NR, int C, int H, int W, int K)
{
    int C2 = 2 * C, Hp = H + 2, Wp = W + 2, HW = H * W;
    int idx = blockIdx.x * blockDim.x + threadIdx.x;
    int total = NR * Hp * Wp * C2;
    if (idx >= total) return;
    int c2 = idx % C2;
    int t  = idx / C2;
    int xp = t % Wp; t /= Wp;
    int yp = t % Hp;
    int ri = t / Hp;
    int iy = yp - 1, ix = xp - 1;
    float v = 0.0f;
    if (iy >= 0 && iy < H && ix >= 0 && ix < W) {
        int hw = iy * W + ix;
        if (c2 < C) v = roi_s[((size_t)ri * C + c2) * HW + hw];
        else        v = tplg[((size_t)(ri / K) * C + (c2 - C)) * HW + hw];
    }
    xh[idx] = (__bf16)v;
}

// ---------------------------------------------------------------------------
// Pre-pack OIHW fp32 weights -> bf16 in the exact per-lane WMMA A layout:
//   wp[((((tm*9 + r)*KC + kc)*32) + lane)*16 + e]
//   A K index (contiguous order, same as B) kA = lh*16 + e;  ci = kc*32 + kA;  co = tm*16+lm
// ---------------------------------------------------------------------------
__global__ void k_pack_w(const float* __restrict__ w, __bf16* __restrict__ wp,
                         int Cout, int Cin)
{
    int idx = blockIdx.x * blockDim.x + threadIdx.x;
    int KC = Cin >> 5;
    int total = Cout * 9 * Cin;
    if (idx >= total) return;
    int e    = idx & 15;
    int lane = (idx >> 4) & 31;
    int rest = idx >> 9;
    int kc   = rest % KC; rest /= KC;
    int r    = rest % 9;
    int tm   = rest / 9;
    int lm = lane & 15, lh = lane >> 4;
    int ci = kc * 32 + (lh << 4) + e;
    int co = tm * 16 + lm;
    wp[idx] = (__bf16)w[((size_t)co * Cin + ci) * 9 + r];
}

// ---------------------------------------------------------------------------
// Implicit-GEMM 3x3 SAME conv, bf16 WMMA, fp32 accumulate.
//   M = C_out, N = Bn*H*W, K = 9*C_in reordered as (r, ci).
// One wave computes a 32(M) x 32(N) macro-tile: 4 accumulators; each A fetch
// is reused across 2 B fetches and vice versa -> 4 WMMA per k-step,
// branch-free body thanks to the zero-padded NHWC input.
// ---------------------------------------------------------------------------
__global__ __launch_bounds__(256)
void k_conv3x3_wmma(const __bf16* __restrict__ xh,   // padded NHWC bf16
                    const __bf16* __restrict__ wp,   // packed bf16
                    const float* __restrict__ bias, float* __restrict__ y,
                    int Bn, int Cin, int H, int W, int Cout, int do_relu)
{
    const int lane = threadIdx.x & 31;
    const int wave = threadIdx.x >> 5;
    const int lm   = lane & 15;
    const int lh   = lane >> 4;
    const int HW   = H * W;
    const int Wp   = W + 2;
    const int Ntot = Bn * HW;
    const int tilesN2 = (Ntot + 31) >> 5;
    const int tilesM2 = Cout >> 5;

    int tile = blockIdx.x * 8 + wave;
    if (tile >= tilesM2 * tilesN2) return;      // uniform per wave
    int tm2 = tile / tilesN2;
    int tn2 = tile - tm2 * tilesN2;

    int  s0 = tn2 * 32 + lm, s1 = s0 + 16;
    bool v0 = s0 < Ntot,     v1 = s1 < Ntot;
    int  ss0 = v0 ? s0 : 0,  ss1 = v1 ? s1 : 0;
    int b0 = ss0 / HW, sp0 = ss0 - b0 * HW, oy0 = sp0 / W, ox0 = sp0 - oy0 * W;
    int b1 = ss1 / HW, sp1 = ss1 - b1 * HW, oy1 = sp1 / W, ox1 = sp1 - oy1 * W;

    const int KC = Cin >> 5;
    const size_t rstep = (size_t)KC * 512;      // packed elems per r-block
    v8f acc00 = {}, acc01 = {}, acc10 = {}, acc11 = {};

    for (int r = 0; r < 9; ++r) {
        int ky = r / 3, kx = r - 3 * ky;
        const __bf16* x0 = xh + ((size_t)(b0 * (H + 2) + oy0 + ky) * Wp + ox0 + kx) * Cin + (lh << 4);
        const __bf16* x1 = xh + ((size_t)(b1 * (H + 2) + oy1 + ky) * Wp + ox1 + kx) * Cin + (lh << 4);
        const __bf16* w0 = wp + ((size_t)((2 * tm2 + 0) * 9 + r) * KC * 32 + lane) * 16;
        const __bf16* w1 = wp + ((size_t)((2 * tm2 + 1) * 9 + r) * KC * 32 + lane) * 16;
        if (r < 8) {                            // prefetch next r-block of weights
            __builtin_prefetch(w0 + rstep, 0, 0);
            __builtin_prefetch(w1 + rstep, 0, 0);
        }
        #pragma unroll 2
        for (int kc = 0; kc < KC; ++kc) {
            v16bf a0  = *(const v16bf*)(w0 + (size_t)kc * 512);
            v16bf a1  = *(const v16bf*)(w1 + (size_t)kc * 512);
            v16bf bm0 = *(const v16bf*)(x0 + (size_t)kc * 32);
            v16bf bm1 = *(const v16bf*)(x1 + (size_t)kc * 32);
            acc00 = __builtin_amdgcn_wmma_f32_16x16x32_bf16(
                        false, a0, false, bm0, (short)0, acc00, false, false);
            acc10 = __builtin_amdgcn_wmma_f32_16x16x32_bf16(
                        false, a1, false, bm0, (short)0, acc10, false, false);
            acc01 = __builtin_amdgcn_wmma_f32_16x16x32_bf16(
                        false, a0, false, bm1, (short)0, acc01, false, false);
            acc11 = __builtin_amdgcn_wmma_f32_16x16x32_bf16(
                        false, a1, false, bm1, (short)0, acc11, false, false);
        }
    }

    #pragma unroll
    for (int v = 0; v < 8; ++v) {
        int co0 = tm2 * 32 + (lh << 3) + v;     // D: VGPR v -> M = v + 8*lh
        int co1 = co0 + 16;
        float b0v = bias[co0], b1v = bias[co1];
        if (v0) {
            float o = acc00[v] + b0v; if (do_relu) o = fmaxf(o, 0.0f);
            y[(size_t)(b0 * Cout + co0) * HW + sp0] = o;
            o = acc10[v] + b1v; if (do_relu) o = fmaxf(o, 0.0f);
            y[(size_t)(b0 * Cout + co1) * HW + sp0] = o;
        }
        if (v1) {
            float o = acc01[v] + b0v; if (do_relu) o = fmaxf(o, 0.0f);
            y[(size_t)(b1 * Cout + co0) * HW + sp1] = o;
            o = acc11[v] + b1v; if (do_relu) o = fmaxf(o, 0.0f);
            y[(size_t)(b1 * Cout + co1) * HW + sp1] = o;
        }
    }
}

// ---------------------------------------------------------------------------
// Thin 1x1 conv (C_out = 1 or 4): plain dot products.
// ---------------------------------------------------------------------------
__global__ void k_conv1x1(const float* __restrict__ x, const float* __restrict__ w,
                          const float* __restrict__ bias, float* __restrict__ y,
                          int Bn, int Cin, int HW, int Cout)
{
    int idx = blockIdx.x * blockDim.x + threadIdx.x;
    int total = Bn * Cout * HW;
    if (idx >= total) return;
    int p  = idx % HW;
    int t  = idx / HW;
    int co = t % Cout;
    int b  = t / Cout;
    const float* xb = x + (size_t)b * Cin * HW + p;
    const float* wr = w + (size_t)co * Cin;
    float acc = bias[co];
    for (int ci = 0; ci < Cin; ++ci) acc += xb[(size_t)ci * HW] * wr[ci];
    y[idx] = acc;
}

// ---------------------------------------------------------------------------
// RPN decode: sigmoid objectness + anchor box decode.
// ---------------------------------------------------------------------------
__global__ void k_rpn_decode(const float* __restrict__ obj, const float* __restrict__ box,
                             float* __restrict__ props, float* __restrict__ scores,
                             int Bn, int H, int W, float stride)
{
    int idx = blockIdx.x * blockDim.x + threadIdx.x;
    int HW = H * W;
    if (idx >= Bn * HW) return;
    int b = idx / HW;
    int s = idx - b * HW;
    int oy = s / W, ox = s - oy * W;

    float d0 = box[((size_t)b * 4 + 0) * HW + s];
    float d1 = box[((size_t)b * 4 + 1) * HW + s];
    float d2 = box[((size_t)b * 4 + 2) * HW + s];
    float d3 = box[((size_t)b * 4 + 3) * HW + s];
    float acx = (ox + 0.5f) * stride;
    float acy = (oy + 0.5f) * stride;
    float cx = acx + d0 * stride;
    float cy = acy + d1 * stride;
    float ww = stride * expf(d2);
    float hh = stride * expf(d3);
    props[(size_t)idx * 4 + 0] = cx - 0.5f * ww;
    props[(size_t)idx * 4 + 1] = cy - 0.5f * hh;
    props[(size_t)idx * 4 + 2] = cx + 0.5f * ww;
    props[(size_t)idx * 4 + 3] = cy + 0.5f * hh;
    scores[idx] = sigmoidf_(obj[(size_t)b * HW + s]);
}

// ---------------------------------------------------------------------------
// Exact top-k via rank counting (ties broken by lower index, matching top_k).
// ---------------------------------------------------------------------------
__global__ void k_topk_rank(const float* __restrict__ scores, const float* __restrict__ props,
                            float* __restrict__ sc_out, float* __restrict__ p_out,
                            int Bn, int N, int k)
{
    int idx = blockIdx.x * blockDim.x + threadIdx.x;
    if (idx >= Bn * N) return;
    int b = idx / N, i = idx - b * N;
    const float* s = scores + (size_t)b * N;
    float si = s[i];
    int rank = 0;
    for (int j = 0; j < N; ++j) {
        float sj = s[j];
        rank += (sj > si) || (sj == si && j < i);
    }
    if (rank < k) {
        sc_out[(size_t)b * k + rank] = si;
        const float* pi = props + ((size_t)b * N + i) * 4;
        float* po = p_out + ((size_t)b * k + rank) * 4;
        po[0] = pi[0]; po[1] = pi[1]; po[2] = pi[2]; po[3] = pi[3];
    }
}

// ---------------------------------------------------------------------------
// NMS: sequential scan over score-sorted boxes, parallel suppression.
// ---------------------------------------------------------------------------
__global__ void k_nms(const float* __restrict__ boxes, int* __restrict__ keep,
                      int Bn, int k, float thr)
{
    extern __shared__ unsigned char supp[];
    int b = blockIdx.x;
    const float* bb = boxes + (size_t)b * k * 4;
    for (int i = threadIdx.x; i < k; i += blockDim.x) supp[i] = 0;
    __syncthreads();
    for (int i = 0; i < k; ++i) {
        bool alive = (supp[i] == 0);
        if (threadIdx.x == 0) keep[(size_t)b * k + i] = alive ? 1 : 0;
        if (alive) {
            float x1 = bb[i * 4 + 0], y1 = bb[i * 4 + 1];
            float x2 = bb[i * 4 + 2], y2 = bb[i * 4 + 3];
            float ai = fmaxf(x2 - x1, 0.0f) * fmaxf(y2 - y1, 0.0f);
            for (int j = i + 1 + threadIdx.x; j < k; j += blockDim.x) {
                float u1 = bb[j * 4 + 0], v1 = bb[j * 4 + 1];
                float u2 = bb[j * 4 + 2], v2 = bb[j * 4 + 3];
                float aj = fmaxf(u2 - u1, 0.0f) * fmaxf(v2 - v1, 0.0f);
                float ix = fmaxf(fminf(x2, u2) - fmaxf(x1, u1), 0.0f);
                float iy = fmaxf(fminf(y2, v2) - fmaxf(y1, v1), 0.0f);
                float inter = ix * iy;
                float iou = inter / fmaxf(ai + aj - inter, 1e-9f);
                if (iou > thr) supp[j] = 1;
            }
        }
        __syncthreads();
    }
}

// ---------------------------------------------------------------------------
// Reorder after NMS: kept boxes first (orig order), suppressed after; take kk.
// ---------------------------------------------------------------------------
__global__ void k_reorder(const float* __restrict__ pk, const float* __restrict__ sck,
                          const int* __restrict__ keep,
                          float* __restrict__ out_p, float* __restrict__ out_s,
                          int Bn, int k, int kk, int out_stride, int out_off)
{
    int idx = blockIdx.x * blockDim.x + threadIdx.x;
    if (idx >= Bn * k) return;
    int b = idx / k, i = idx - b * k;
    const int* kp = keep + (size_t)b * k;
    int kept_before = 0, total_kept = 0;
    for (int j = 0; j < k; ++j) {
        int kj = kp[j];
        total_kept += kj;
        if (j < i) kept_before += kj;
    }
    int ki = kp[i];
    int pos = ki ? kept_before : (total_kept + i - kept_before);
    if (pos >= kk) return;
    size_t row = (size_t)b * out_stride + out_off + pos;
    if (ki) {
        const float* pi = pk + ((size_t)b * k + i) * 4;
        out_p[row * 4 + 0] = pi[0]; out_p[row * 4 + 1] = pi[1];
        out_p[row * 4 + 2] = pi[2]; out_p[row * 4 + 3] = pi[3];
        out_s[row] = sck[(size_t)b * k + i];
    } else {
        out_p[row * 4 + 0] = 0.0f; out_p[row * 4 + 1] = 0.0f;
        out_p[row * 4 + 2] = 0.0f; out_p[row * 4 + 3] = 0.0f;
        out_s[row] = -1.0f;
    }
}

// ---------------------------------------------------------------------------
// Emit props_t / sc_t (first K of the Kf-sorted list) into d_out.
// ---------------------------------------------------------------------------
__global__ void k_emit(const float* __restrict__ pKf, const float* __restrict__ sKf,
                       float* __restrict__ out, int Bn, int Kf, int K)
{
    int idx = blockIdx.x * blockDim.x + threadIdx.x;
    if (idx >= Bn * K) return;
    int b = idx / K, kq = idx - b * K;
    const float* pi = pKf + ((size_t)b * Kf + kq) * 4;
    out[(size_t)idx * 4 + 0] = pi[0];
    out[(size_t)idx * 4 + 1] = pi[1];
    out[(size_t)idx * 4 + 2] = pi[2];
    out[(size_t)idx * 4 + 3] = pi[3];
    out[(size_t)Bn * K * 4 + idx] = sKf[(size_t)b * Kf + kq];
}

// ---------------------------------------------------------------------------
// ROI align (7x7, sampling ratio 2), reference bilinear/valid semantics.
// ---------------------------------------------------------------------------
__device__ __forceinline__ float bilin_(const float* img, int H, int W, float px, float py)
{
    bool valid = (px > -1.0f) && (px < (float)W) && (py > -1.0f) && (py < (float)H);
    px = fminf(fmaxf(px, 0.0f), (float)W - 1.0f);
    py = fminf(fmaxf(py, 0.0f), (float)H - 1.0f);
    float x0 = floorf(px), y0 = floorf(py);
    float x1 = fminf(x0 + 1.0f, (float)W - 1.0f);
    float y1 = fminf(y0 + 1.0f, (float)H - 1.0f);
    float lx = px - x0, ly = py - y0;
    int x0i = (int)x0, x1i = (int)x1, y0i = (int)y0, y1i = (int)y1;
    float v00 = img[y0i * W + x0i], v01 = img[y0i * W + x1i];
    float v10 = img[y1i * W + x0i], v11 = img[y1i * W + x1i];
    float o = v00 * (1 - ly) * (1 - lx) + v01 * (1 - ly) * lx
            + v10 * ly * (1 - lx)       + v11 * ly * lx;
    return valid ? o : 0.0f;
}

__global__ void k_roi_align(const float* __restrict__ feat, const float* __restrict__ boxes,
                            float* __restrict__ out, int nroi, int rois_per_batch,
                            int C, int H, int W, float scale)
{
    int idx = blockIdx.x * blockDim.x + threadIdx.x;
    int total = nroi * C * 49;
    if (idx >= total) return;
    int oxy = idx % 49;
    int t   = idx / 49;
    int c   = t % C;
    int ri  = t / C;
    int oy = oxy / 7, ox = oxy - oy * 7;
    int bimg = rois_per_batch ? (ri / rois_per_batch) : ri;

    const float* box = boxes + (size_t)ri * 4;
    float x1 = box[0] * scale - 0.5f;
    float y1 = box[1] * scale - 0.5f;
    float x2 = box[2] * scale - 0.5f;
    float y2 = box[3] * scale - 0.5f;
    float bw = (x2 - x1) / 7.0f;
    float bh = (y2 - y1) / 7.0f;
    const float* img = feat + ((size_t)bimg * C + c) * H * W;

    float acc = 0.0f;
    #pragma unroll
    for (int sy = 0; sy < 2; ++sy)
        #pragma unroll
        for (int sx = 0; sx < 2; ++sx) {
            float px = x1 + ((ox * 2 + sx) + 0.5f) * 0.5f * bw;
            float py = y1 + ((oy * 2 + sy) + 0.5f) * 0.5f * bh;
            acc += bilin_(img, H, W, px, py);
        }
    out[idx] = acc * 0.25f;
}

// ---------------------------------------------------------------------------
// Global average pool over HW per (n, c);  x layout [n][c][hw].
// ---------------------------------------------------------------------------
__global__ void k_gap(const float* __restrict__ x, float* __restrict__ out,
                      int N, int C, int HW)
{
    int idx = blockIdx.x * blockDim.x + threadIdx.x;
    if (idx >= N * C) return;
    const float* p = x + (size_t)idx * HW;
    float a = 0.0f;
    for (int i = 0; i < HW; ++i) a += p[i];
    out[idx] = a / (float)HW;
}

// ---------------------------------------------------------------------------
// qg3 channel-gating MLP.
// ---------------------------------------------------------------------------
__global__ void k_qg_mlp(const float* __restrict__ gatew,
                         const float* __restrict__ fc1w, const float* __restrict__ fc1b,
                         const float* __restrict__ fc2w, const float* __restrict__ fc2b,
                         float* __restrict__ gate, int C, int hid)
{
    __shared__ float hbuf[64];
    int b = blockIdx.x;
    const float* g = gatew + (size_t)b * C;
    if ((int)threadIdx.x < hid) {
        float a = fc1b[threadIdx.x];
        for (int c = 0; c < C; ++c) a += fc1w[(size_t)threadIdx.x * C + c] * g[c];
        hbuf[threadIdx.x] = fmaxf(a, 0.0f);
    }
    __syncthreads();
    for (int c = threadIdx.x; c < C; c += blockDim.x) {
        float a = fc2b[c];
        for (int h = 0; h < hid; ++h) a += fc2w[(size_t)c * hid + h] * hbuf[h];
        gate[(size_t)b * C + c] = sigmoidf_(a);
    }
}

__global__ void k_scale(const float* __restrict__ x, const float* __restrict__ gate,
                        float* __restrict__ y, int Bn, int C, int HW)
{
    int idx = blockIdx.x * blockDim.x + threadIdx.x;
    if (idx >= Bn * C * HW) return;
    int c = (idx / HW) % C;
    int b = idx / (C * HW);
    y[idx] = x[idx] * gate[(size_t)b * C + c];
}

// ---------------------------------------------------------------------------
// Final head: sem / sim / bkg / fused / delta written straight to d_out.
// ---------------------------------------------------------------------------
__global__ void k_head_out(const float* __restrict__ gap_h, const float* __restrict__ gap_sh,
                           const float* __restrict__ gatew,
                           const float* __restrict__ cls_w,  const float* __restrict__ cls_b,
                           const float* __restrict__ simw,   const float* __restrict__ simb,
                           const float* __restrict__ reg_w,  const float* __restrict__ reg_b,
                           float* __restrict__ out, int Bn, int K, int C)
{
    int ri = blockIdx.x * blockDim.x + threadIdx.x;
    int NK = Bn * K;
    if (ri >= NK) return;
    int b = ri / K;
    const float* gh = gap_h  + (size_t)ri * C;
    const float* gs = gap_sh + (size_t)ri * C;

    float sem = cls_b[0];
    float sim = simb[0];
    for (int c = 0; c < C; ++c) { sem += cls_w[c] * gh[c]; sim += simw[c] * gs[c]; }
    sem = sigmoidf_(sem);
    sim = sigmoidf_(sim);

    float gm = 0.0f;
    for (int c = 0; c < C; ++c) gm += gatew[(size_t)b * C + c];
    float bkg = sigmoidf_(gm / (float)C);
    float fused = sigmoidf_(0.5f * sem + 0.4f * sim + 0.1f * bkg);

    out[(size_t)NK * 5 + ri] = fused;
    out[(size_t)NK * 6 + ri] = sem;
    out[(size_t)NK * 7 + ri] = sim;
    out[(size_t)NK * 8 + ri] = bkg;
    #pragma unroll
    for (int o = 0; o < 4; ++o) {
        float d = reg_b[o];
        for (int c = 0; c < C; ++c) d += reg_w[(size_t)o * C + c] * gh[c];
        out[(size_t)NK * 9 + (size_t)ri * 4 + o] = d;
    }
}

// ---------------------------------------------------------------------------
// Orchestration
// ---------------------------------------------------------------------------
extern "C" void kernel_launch(void* const* d_in, const int* in_sizes, int n_in,
                              void* d_out, int out_size, void* d_ws, size_t ws_size,
                              hipStream_t stream)
{
    (void)n_in; (void)out_size; (void)ws_size;
    const float* p3   = (const float*)d_in[0];
    const float* p4   = (const float*)d_in[1];
    const float* p5   = (const float*)d_in[2];
    const float* tbox = (const float*)d_in[3];
    const int Bn = in_sizes[3] / 4;

    auto P = [&](int i) { return (const float*)d_in[4 + i]; };
    const float* qg3_fc1w = P(24); const float* qg3_fc1b = P(25);
    const float* qg3_fc2w = P(26); const float* qg3_fc2b = P(27);
    const float* sh_w1 = P(36); const float* sh_b1 = P(37);
    const float* sh_w2 = P(38); const float* sh_b2 = P(39);
    const float* cls_w = P(40); const float* cls_b = P(41);
    const float* simc_w1 = P(42); const float* simc_b1 = P(43);
    const float* simc_w2 = P(44); const float* simc_b2 = P(45);
    const float* simfc_w = P(46); const float* simfc_b = P(47);
    const float* reg_w   = P(48); const float* reg_b   = P(49);

    const int Cch[3]    = {256, 512, 1024};
    const int Hs[3]     = {80, 40, 20};
    const int strd[3]   = {8, 16, 32};
    const int lvlOff[3] = {0, 1000, 2000};
    const int CAT = 2400;
    const int Kf = 1000, Kt = 300, NR = Bn * Kt;

    // ---- workspace bump allocator ----
    size_t off = 0;
    auto alloc = [&](size_t nbytes) {
        void* p = (char*)d_ws + off;
        off += ((nbytes + 255) / 256) * 256;
        return p;
    };
    float*  bufA    = (float*)alloc((size_t)Bn * 256 * 6400 * 4);
    float*  bufB    = (float*)alloc((size_t)Bn * 256 * 6400 * 4);
    __bf16* xh      = (__bf16*)alloc((size_t)NR * 81 * 512 * 2);       // max padded NHWC bf16
    __bf16* wpk     = (__bf16*)alloc((size_t)1024 * 1024 * 9 * 2);     // max packed W
    float*  objb    = (float*)alloc((size_t)Bn * 6400 * 4);
    float*  boxb    = (float*)alloc((size_t)Bn * 4 * 6400 * 4);
    float*  propsL  = (float*)alloc((size_t)Bn * 6400 * 4 * 4);
    float*  scoresL = (float*)alloc((size_t)Bn * 6400 * 4);
    float*  pk      = (float*)alloc((size_t)Bn * 3000 * 4 * 4);
    float*  sck     = (float*)alloc((size_t)Bn * 3000 * 4);
    int*    keepb   = (int*)alloc((size_t)Bn * 3000 * 4);
    float*  propsAll= (float*)alloc((size_t)Bn * CAT * 4 * 4);
    float*  scAll   = (float*)alloc((size_t)Bn * CAT * 4);
    float*  propsKf = (float*)alloc((size_t)Bn * Kf * 4 * 4);
    float*  scKf    = (float*)alloc((size_t)Bn * Kf * 4);
    float*  tpl3    = (float*)alloc((size_t)Bn * 256 * 49 * 4);
    float*  gatew   = (float*)alloc((size_t)Bn * 256 * 4);
    float*  gate    = (float*)alloc((size_t)Bn * 256 * 4);
    float*  p3g     = (float*)alloc((size_t)Bn * 256 * 6400 * 4);
    float*  tpl3g   = (float*)alloc((size_t)Bn * 256 * 49 * 4);
    float*  roi_s   = (float*)alloc((size_t)NR * 256 * 49 * 4);
    float*  h1      = (float*)alloc((size_t)NR * 256 * 49 * 4);
    float*  h2      = (float*)alloc((size_t)NR * 256 * 49 * 4);
    float*  sh1 = h1;     // reuse: h1 dead after h2
    float*  sh2 = roi_s;  // reuse: roi_s dead after cat
    float*  gapH    = (float*)alloc((size_t)NR * 256 * 4);
    float*  gapS    = (float*)alloc((size_t)NR * 256 * 4);

    float* out = (float*)d_out;
    const float* feats[3] = {p3, p4, p5};

    // ---- RPN levels ----
    for (int l = 0; l < 3; ++l) {
        int C = Cch[l], H = Hs[l], W = Hs[l], HW = H * W;
        const float* w1 = P(8 * l + 0); const float* b1 = P(8 * l + 1);
        const float* w2 = P(8 * l + 2); const float* b2 = P(8 * l + 3);
        const float* ow = P(8 * l + 4); const float* ob = P(8 * l + 5);
        const float* bw = P(8 * l + 6); const float* bb = P(8 * l + 7);

        int Ntot  = Bn * HW;
        int padel = Bn * (H + 2) * (W + 2) * C;
        int wtot  = C * C * 9;
        int tiles = (C / 32) * ((Ntot + 31) / 32);

        k_to_nhwc_pad<<<(padel + 255) / 256, 256, 0, stream>>>(feats[l], xh, Bn, C, H, W);
        k_pack_w<<<(wtot + 255) / 256, 256, 0, stream>>>(w1, wpk, C, C);
        k_conv3x3_wmma<<<(tiles + 7) / 8, 256, 0, stream>>>(xh, wpk, b1, bufA, Bn, C, H, W, C, 1);

        k_to_nhwc_pad<<<(padel + 255) / 256, 256, 0, stream>>>(bufA, xh, Bn, C, H, W);
        k_pack_w<<<(wtot + 255) / 256, 256, 0, stream>>>(w2, wpk, C, C);
        k_conv3x3_wmma<<<(tiles + 7) / 8, 256, 0, stream>>>(xh, wpk, b2, bufB, Bn, C, H, W, C, 1);

        k_conv1x1<<<(Bn * HW + 255) / 256, 256, 0, stream>>>(bufB, ow, ob, objb, Bn, C, HW, 1);
        k_conv1x1<<<(Bn * 4 * HW + 255) / 256, 256, 0, stream>>>(bufB, bw, bb, boxb, Bn, C, HW, 4);
        k_rpn_decode<<<(Bn * HW + 255) / 256, 256, 0, stream>>>(objb, boxb, propsL, scoresL,
                                                                Bn, H, W, (float)strd[l]);
        int kl = HW < 3000 ? HW : 3000;
        k_topk_rank<<<(Bn * HW + 255) / 256, 256, 0, stream>>>(scoresL, propsL, sck, pk, Bn, HW, kl);
        k_nms<<<Bn, 256, kl, stream>>>(pk, keepb, Bn, kl, 0.7f);
        int kk = kl < 1000 ? kl : 1000;
        k_reorder<<<(Bn * kl + 255) / 256, 256, 0, stream>>>(pk, sck, keepb, propsAll, scAll,
                                                             Bn, kl, kk, CAT, lvlOff[l]);
    }

    // ---- combined top-Kf, emit props_t / sc_t ----
    k_topk_rank<<<(Bn * CAT + 255) / 256, 256, 0, stream>>>(scAll, propsAll, scKf, propsKf, Bn, CAT, Kf);
    k_emit<<<(Bn * Kt + 255) / 256, 256, 0, stream>>>(propsKf, scKf, out, Bn, Kf, Kt);

    // ---- template gating (qg4/qg5 are dead in the reference: skipped) ----
    k_roi_align<<<(Bn * 256 * 49 + 255) / 256, 256, 0, stream>>>(p3, tbox, tpl3, Bn, 0, 256, 80, 80, 0.125f);
    k_gap<<<(Bn * 256 + 255) / 256, 256, 0, stream>>>(tpl3, gatew, Bn, 256, 49);
    k_qg_mlp<<<Bn, 256, 0, stream>>>(gatew, qg3_fc1w, qg3_fc1b, qg3_fc2w, qg3_fc2b, gate, 256, 16);
    k_scale<<<(Bn * 256 * 6400 + 255) / 256, 256, 0, stream>>>(p3, gate, p3g, Bn, 256, 6400);

    // ---- ROI features on gated p3 ----
    k_roi_align<<<(Bn * 256 * 49 + 255) / 256, 256, 0, stream>>>(p3g, tbox, tpl3g, Bn, 0, 256, 80, 80, 0.125f);
    k_roi_align<<<(NR * 256 * 49 + 255) / 256, 256, 0, stream>>>(p3g, out /*props_t*/, roi_s,
                                                                 NR, Kt, 256, 80, 80, 0.125f);

    // ---- head convs (WMMA, N = NR*49) ----
    int NtotH  = NR * 49;
    int tilesH = (256 / 32) * ((NtotH + 31) / 32);
    int padelH = NR * 81 * 256;

    k_to_nhwc_pad<<<(padelH + 255) / 256, 256, 0, stream>>>(roi_s, xh, NR, 256, 7, 7);
    k_pack_w<<<(256 * 256 * 9 + 255) / 256, 256, 0, stream>>>(sh_w1, wpk, 256, 256);
    k_conv3x3_wmma<<<(tilesH + 7) / 8, 256, 0, stream>>>(xh, wpk, sh_b1, h1, NR, 256, 7, 7, 256, 1);

    k_to_nhwc_pad<<<(padelH + 255) / 256, 256, 0, stream>>>(h1, xh, NR, 256, 7, 7);
    k_pack_w<<<(256 * 256 * 9 + 255) / 256, 256, 0, stream>>>(sh_w2, wpk, 256, 256);
    k_conv3x3_wmma<<<(tilesH + 7) / 8, 256, 0, stream>>>(xh, wpk, sh_b2, h2, NR, 256, 7, 7, 256, 1);

    k_cat_nhwc_pad<<<(NR * 81 * 512 + 255) / 256, 256, 0, stream>>>(roi_s, tpl3g, xh, NR, 256, 7, 7, Kt);
    k_pack_w<<<(256 * 512 * 9 + 255) / 256, 256, 0, stream>>>(simc_w1, wpk, 256, 512);
    k_conv3x3_wmma<<<(tilesH + 7) / 8, 256, 0, stream>>>(xh, wpk, simc_b1, sh1, NR, 512, 7, 7, 256, 1);

    k_to_nhwc_pad<<<(padelH + 255) / 256, 256, 0, stream>>>(sh1, xh, NR, 256, 7, 7);
    k_pack_w<<<(256 * 256 * 9 + 255) / 256, 256, 0, stream>>>(simc_w2, wpk, 256, 256);
    k_conv3x3_wmma<<<(tilesH + 7) / 8, 256, 0, stream>>>(xh, wpk, simc_b2, sh2, NR, 256, 7, 7, 256, 1);

    // ---- pooled heads + fusion ----
    k_gap<<<(NR * 256 + 255) / 256, 256, 0, stream>>>(h2, gapH, NR, 256, 49);
    k_gap<<<(NR * 256 + 255) / 256, 256, 0, stream>>>(sh2, gapS, NR, 256, 49);
    k_head_out<<<(NR + 255) / 256, 256, 0, stream>>>(gapH, gapS, gatew, cls_w, cls_b,
                                                     simfc_w, simfc_b, reg_w, reg_b,
                                                     out, Bn, Kt, 256);
}
